// Net_88536455840449
// MI455X (gfx1250) — matrix operand order repaired and checked by reading, not compiled, not run on hardware
//
#include <hip/hip_runtime.h>
#include <math.h>

// ---------------------------------------------------------------------------
// GAT forward for MI455X (gfx1250, wave32).
// GEMMs run on v_wmma_f32_16x16x32_bf16; attention/aggregation is the
// memory-bound part and uses coalesced gathers + fp32 atomics.
// ---------------------------------------------------------------------------

#define HH 16            // heads
#define BN_EPS 1e-5f

typedef __bf16 bf16_t;
typedef __attribute__((ext_vector_type(16))) __bf16 v16bf;
typedef __attribute__((ext_vector_type(8)))  __bf16 v8bf;
typedef __attribute__((ext_vector_type(8)))  float  v8f;

union AFrag { v16bf v; struct { v8bf lo, hi; } h; };
union BfBits { unsigned short u; __bf16 b; };

__device__ __forceinline__ bf16_t f2bf(float f) {
  BfBits r;
  unsigned u = __float_as_uint(f);
  r.u = (unsigned short)((u + 0x7FFFu + ((u >> 16) & 1u)) >> 16);   // RNE
  return r.b;
}

__device__ __forceinline__ void atomic_max_f32(float* addr, float val) {
  // IEEE trick: positive floats order as signed ints, negatives as unsigned.
  if (val >= 0.0f) atomicMax((int*)addr, __float_as_int(val));
  else             atomicMin((unsigned int*)addr, __float_as_uint(val));
}

// ------------------------------ fill ---------------------------------------
__global__ void k_fill(float* __restrict__ p, int n, float v) {
  int t = blockIdx.x * blockDim.x + threadIdx.x;
  if (t < n) p[t] = v;
}

// ----------------- BN (eval) + convert to bf16 GEMM input ------------------
__global__ void k_bn_in_bf16(const float* __restrict__ x,
                             const float* __restrict__ g, const float* __restrict__ b,
                             const float* __restrict__ m, const float* __restrict__ v,
                             bf16_t* __restrict__ out, int n_elems, int C) {
  int t = blockIdx.x * blockDim.x + threadIdx.x;
  if (t >= n_elems) return;
  int c = t % C;
  float val = (x[t] - m[c]) * rsqrtf(v[c] + BN_EPS) * g[c] + b[c];
  out[t] = f2bf(val);
}

// --------- transpose W[K_in][Out] -> Wt[Out][Kpad] bf16 (zero-padded K) ----
__global__ void k_transpose_w(const float* __restrict__ W, bf16_t* __restrict__ Wt,
                              int K_in, int Out, int Kpad) {
  int t = blockIdx.x * blockDim.x + threadIdx.x;
  if (t >= Out * Kpad) return;
  int o = t / Kpad, k = t % Kpad;
  Wt[t] = (k < K_in) ? f2bf(W[(size_t)k * Out + o]) : f2bf(0.0f);
}

// ---------------- WMMA bf16 GEMM: C[M][Ncols] = A[M][K] * Bt[Ncols][K]^T ---
// One wave per block; wave computes a 16(M) x 32(N) tile, K in steps of 32.
__global__ void k_gemm_bf16_wmma(const bf16_t* __restrict__ A,
                                 const bf16_t* __restrict__ Bt,
                                 float* __restrict__ C,
                                 int M, int Ncols, int K) {
  const int lane = threadIdx.x;          // 0..31
  const int mt   = blockIdx.x * 16;
  const int nt   = blockIdx.y * 32;
  const int l15  = lane & 15;
  const int half = lane >> 4;            // 0 or 1

  v8f acc0 = {}; v8f acc1 = {};
  const bf16_t* arow = A  + (size_t)(mt + l15) * K;
  const bf16_t* bc0  = Bt + (size_t)(nt + l15) * K;        // col nt+l15
  const bf16_t* bc1  = Bt + (size_t)(nt + 16 + l15) * K;   // col nt+16+l15

  for (int k0 = 0; k0 < K; k0 += 32) {
    AFrag a;
    a.h.lo = *(const v8bf*)(arow + k0 + half * 16);         // K k0+half*16..+7
    a.h.hi = *(const v8bf*)(arow + k0 + half * 16 + 8);     // K k0+half*16+8..+15
    v16bf b0 = *(const v16bf*)(bc0 + k0 + half * 16);       // K k0+half*16..+15
    v16bf b1 = *(const v16bf*)(bc1 + k0 + half * 16);
    if (k0 + 32 < K) __builtin_prefetch(arow + k0 + 32, 0, 3);
    acc0 = __builtin_amdgcn_wmma_f32_16x16x32_bf16(false, a.v, false, b0,
                                                   (short)0, acc0, false, false);
    acc1 = __builtin_amdgcn_wmma_f32_16x16x32_bf16(false, a.v, false, b1,
                                                   (short)0, acc1, false, false);
  }
#pragma unroll
  for (int r = 0; r < 8; ++r) {
    int mrow = mt + r + half * 8;                            // C/D VGPR layout
    C[(size_t)mrow * Ncols + nt + l15]      = acc0[r];
    C[(size_t)mrow * Ncols + nt + 16 + l15] = acc1[r];
  }
}

// ------------- per-node attention logits: alo/ahi [N][H] -------------------
__global__ void k_attn_logits(const float* __restrict__ xt,
                              const float* __restrict__ asrc, const float* __restrict__ adst,
                              float* __restrict__ alo, float* __restrict__ ahi,
                              int Nn, int C) {
  int t = blockIdx.x * blockDim.x + threadIdx.x;
  if (t >= Nn * HH) return;
  int n = t >> 4, h = t & 15;
  const float* xp = xt + ((size_t)n * HH + h) * C;
  const float* ap = asrc + h * C;
  const float* bp = adst + h * C;
  float s0 = 0.f, s1 = 0.f;
  for (int c = 0; c < C; ++c) { float v = xp[c]; s0 += v * ap[c]; s1 += v * bp[c]; }
  alo[t] = s0; ahi[t] = s1;
}

// --------------- per-edge alpha + segment max (incl. self loops) -----------
__global__ void k_edge_max(const int* __restrict__ ei, int E, int Nn,
                           const float* __restrict__ alo, const float* __restrict__ ahi,
                           float* __restrict__ wbuf, float* __restrict__ mx) {
  int t = blockIdx.x * blockDim.x + threadIdx.x;
  int EP = E + Nn;
  if (t >= EP * HH) return;
  int e = t >> 4, h = t & 15;
  int src = (e < E) ? ei[e]     : (e - E);
  int dst = (e < E) ? ei[E + e] : (e - E);
  float a = alo[src * HH + h] + ahi[dst * HH + h];
  a = (a > 0.f) ? a : 0.2f * a;                 // leaky_relu slope 0.2
  wbuf[t] = a;
  atomic_max_f32(&mx[dst * HH + h], a);
}

// --------------- per-edge exp(alpha - max) + segment sum -------------------
__global__ void k_edge_exp(const int* __restrict__ ei, int E, int Nn,
                           float* __restrict__ wbuf, const float* __restrict__ mx,
                           float* __restrict__ s) {
  int t = blockIdx.x * blockDim.x + threadIdx.x;
  int EP = E + Nn;
  if (t >= EP * HH) return;
  int e = t >> 4, h = t & 15;
  int dst = (e < E) ? ei[E + e] : (e - E);
  float ee = __expf(wbuf[t] - mx[dst * HH + h]);
  wbuf[t] = ee;
  atomicAdd(&s[dst * HH + h], ee);
}

// --- weighted aggregate: one wave32 per edge; head weights via shuffles ----
__global__ void k_edge_aggregate(const int* __restrict__ ei, int E, int Nn,
                                 const float* __restrict__ wbuf, const float* __restrict__ s,
                                 const float* __restrict__ xt, float* __restrict__ out,
                                 int C) {
  int wave = (blockIdx.x * blockDim.x + threadIdx.x) >> 5;
  int lane = threadIdx.x & 31;
  int EP = E + Nn;
  if (wave >= EP) return;
  int e = wave;
  int src = (e < E) ? ei[e]     : (e - E);
  int dst = (e < E) ? ei[E + e] : (e - E);
  float w_l = 0.f;
  if (lane < HH) {
    float ee = wbuf[e * HH + lane];
    float ss = s[dst * HH + lane];
    w_l = ee / (ss + 1e-16f) * (1.0f / (float)HH);   // fold head-mean in
  }
  float wh[HH];
#pragma unroll
  for (int h = 0; h < HH; ++h) wh[h] = __shfl(w_l, h, 32);
  const float* xs = xt + (size_t)src * (HH * C);
  for (int c = lane; c < C; c += 32) {
    float acc = 0.f;
#pragma unroll
    for (int h = 0; h < HH; ++h) acc += wh[h] * xs[h * C + c];
    atomicAdd(&out[(size_t)dst * C + c], acc);
  }
}

// ----- layer-1 epilogue: +bias, leaky(0.01), BN2, bf16 + zero-pad K --------
__global__ void k_post1(const float* __restrict__ out1, const float* __restrict__ bias,
                        const float* __restrict__ g, const float* __restrict__ b,
                        const float* __restrict__ m, const float* __restrict__ v,
                        bf16_t* __restrict__ A2, int Nn, int C, int Kpad) {
  int t = blockIdx.x * blockDim.x + threadIdx.x;
  if (t >= Nn * Kpad) return;
  int n = t / Kpad, c = t % Kpad;
  float val = 0.f;
  if (c < C) {
    float a = out1[(size_t)n * C + c] + bias[c];
    a = (a > 0.f) ? a : 0.01f * a;
    val = (a - m[c]) * rsqrtf(v[c] + BN_EPS) * g[c] + b[c];
  }
  A2[t] = f2bf(val);
}

// ----- layer-2 epilogue: +bias, leaky(0.01), mean-pool accumulation --------
__global__ void k_post2(const float* __restrict__ out2, const float* __restrict__ bias,
                        const int* __restrict__ batch,
                        float* __restrict__ pooled, float* __restrict__ cnt,
                        int Nn, int C) {
  int t = blockIdx.x * blockDim.x + threadIdx.x;
  if (t >= Nn * C) return;
  int n = t / C, c = t % C;
  float a = out2[t] + bias[c];
  a = (a > 0.f) ? a : 0.01f * a;
  int gI = batch[n];
  atomicAdd(&pooled[(size_t)gI * C + c], a);
  if (c == 0) atomicAdd(&cnt[gI], 1.0f);
}

// -------------------------- final FC [G] -> d_out --------------------------
__global__ void k_final(const float* __restrict__ pooled, const float* __restrict__ cnt,
                        const float* __restrict__ fw, const float* __restrict__ fb,
                        float* __restrict__ out, int G, int C) {
  int gI = blockIdx.x * blockDim.x + threadIdx.x;
  if (gI >= G) return;
  float cc = cnt[gI]; cc = (cc > 1.f) ? cc : 1.f;
  float acc = fb[0];
  for (int c = 0; c < C; ++c) acc += pooled[(size_t)gI * C + c] / cc * fw[c];
  out[gI] = acc;
}

// ---------------------------------------------------------------------------
extern "C" void kernel_launch(void* const* d_in, const int* in_sizes, int n_in,
                              void* d_out, int out_size, void* d_ws, size_t ws_size,
                              hipStream_t stream) {
  (void)n_in; (void)out_size; (void)ws_size;
  const float* x    = (const float*)d_in[0];
  const int*   ei   = (const int*)d_in[1];
  const int*   bat  = (const int*)d_in[2];
  const float* bn1g = (const float*)d_in[3];
  const float* bn1b = (const float*)d_in[4];
  const float* bn1m = (const float*)d_in[5];
  const float* bn1v = (const float*)d_in[6];
  const float* W1   = (const float*)d_in[7];
  const float* as1  = (const float*)d_in[8];
  const float* ad1  = (const float*)d_in[9];
  const float* b1   = (const float*)d_in[10];
  const float* bn2g = (const float*)d_in[11];
  const float* bn2b = (const float*)d_in[12];
  const float* bn2m = (const float*)d_in[13];
  const float* bn2v = (const float*)d_in[14];
  const float* W2   = (const float*)d_in[15];
  const float* as2  = (const float*)d_in[16];
  const float* ad2  = (const float*)d_in[17];
  const float* b2   = (const float*)d_in[18];
  const float* fcw  = (const float*)d_in[19];
  const float* fcb  = (const float*)d_in[20];
  float* outp = (float*)d_out;

  const int F_IN = 128, C1 = 60, C2 = 30, G = 256;
  const int N  = in_sizes[0] / F_IN;          // 50000 (multiple of 16)
  const int E  = in_sizes[1] / 2;             // 200000
  const int EP = E + N;                       // + self loops
  const int HC1 = HH * C1;                    // 960
  const int HC2 = HH * C2;                    // 480
  const int K2  = 64;                         // C1 zero-padded for WMMA K

  // ---- workspace layout (~383 MB) ----
  char* base = (char*)d_ws; size_t off = 0;
  auto alloc = [&](size_t bytes) -> void* {
    void* p = base + off; off = (off + bytes + 255) & ~(size_t)255; return p;
  };
  bf16_t* W1t  = (bf16_t*)alloc((size_t)HC1 * F_IN * 2);
  bf16_t* A1   = (bf16_t*)alloc((size_t)N * F_IN * 2);
  float*  xt1  = (float*)alloc((size_t)N * HC1 * 4);
  float*  alo1 = (float*)alloc((size_t)N * HH * 4);
  float*  ahi1 = (float*)alloc((size_t)N * HH * 4);
  float*  mx1  = (float*)alloc((size_t)N * HH * 4);
  float*  s1   = (float*)alloc((size_t)N * HH * 4);
  float*  wb1  = (float*)alloc((size_t)EP * HH * 4);
  float*  out1 = (float*)alloc((size_t)N * C1 * 4);
  bf16_t* W2t  = (bf16_t*)alloc((size_t)HC2 * K2 * 2);
  bf16_t* A2   = (bf16_t*)alloc((size_t)N * K2 * 2);
  float*  xt2  = (float*)alloc((size_t)N * HC2 * 4);
  float*  alo2 = (float*)alloc((size_t)N * HH * 4);
  float*  ahi2 = (float*)alloc((size_t)N * HH * 4);
  float*  mx2  = (float*)alloc((size_t)N * HH * 4);
  float*  s2   = (float*)alloc((size_t)N * HH * 4);
  float*  wb2  = (float*)alloc((size_t)EP * HH * 4);
  float*  out2 = (float*)alloc((size_t)N * C2 * 4);
  float*  pooled = (float*)alloc((size_t)G * C2 * 4);
  float*  cnt    = (float*)alloc((size_t)G * 4);

  const int TB = 256;
  auto nb = [&](long long n) { return dim3((unsigned)((n + TB - 1) / TB)); };

  // ---- re-initialize all accumulators (fresh every call) ----
  k_fill<<<nb((long long)N * HH), TB, 0, stream>>>(mx1, N * HH, -INFINITY);
  k_fill<<<nb((long long)N * HH), TB, 0, stream>>>(s1,  N * HH, 0.f);
  k_fill<<<nb((long long)N * C1), TB, 0, stream>>>(out1, N * C1, 0.f);
  k_fill<<<nb((long long)N * HH), TB, 0, stream>>>(mx2, N * HH, -INFINITY);
  k_fill<<<nb((long long)N * HH), TB, 0, stream>>>(s2,  N * HH, 0.f);
  k_fill<<<nb((long long)N * C2), TB, 0, stream>>>(out2, N * C2, 0.f);
  k_fill<<<nb(G * C2), TB, 0, stream>>>(pooled, G * C2, 0.f);
  k_fill<<<nb(G), TB, 0, stream>>>(cnt, G, 0.f);

  // ---- weight prep (bf16, transposed to [out][K]) ----
  k_transpose_w<<<nb((long long)HC1 * F_IN), TB, 0, stream>>>(W1, W1t, F_IN, HC1, F_IN);
  k_transpose_w<<<nb((long long)HC2 * K2), TB, 0, stream>>>(W2, W2t, C1, HC2, K2);

  // ---- layer 1 ----
  k_bn_in_bf16<<<nb((long long)N * F_IN), TB, 0, stream>>>(x, bn1g, bn1b, bn1m, bn1v,
                                                           A1, N * F_IN, F_IN);
  k_gemm_bf16_wmma<<<dim3(N / 16, HC1 / 32), 32, 0, stream>>>(A1, W1t, xt1, N, HC1, F_IN);
  k_attn_logits<<<nb((long long)N * HH), TB, 0, stream>>>(xt1, as1, ad1, alo1, ahi1, N, C1);
  k_edge_max<<<nb((long long)EP * HH), TB, 0, stream>>>(ei, E, N, alo1, ahi1, wb1, mx1);
  k_edge_exp<<<nb((long long)EP * HH), TB, 0, stream>>>(ei, E, N, wb1, mx1, s1);
  k_edge_aggregate<<<nb((long long)EP * 32), TB, 0, stream>>>(ei, E, N, wb1, s1, xt1, out1, C1);
  k_post1<<<nb((long long)N * K2), TB, 0, stream>>>(out1, b1, bn2g, bn2b, bn2m, bn2v,
                                                    A2, N, C1, K2);

  // ---- layer 2 ----
  k_gemm_bf16_wmma<<<dim3(N / 16, HC2 / 32), 32, 0, stream>>>(A2, W2t, xt2, N, HC2, K2);
  k_attn_logits<<<nb((long long)N * HH), TB, 0, stream>>>(xt2, as2, ad2, alo2, ahi2, N, C2);
  k_edge_max<<<nb((long long)EP * HH), TB, 0, stream>>>(ei, E, N, alo2, ahi2, wb2, mx2);
  k_edge_exp<<<nb((long long)EP * HH), TB, 0, stream>>>(ei, E, N, wb2, mx2, s2);
  k_edge_aggregate<<<nb((long long)EP * 32), TB, 0, stream>>>(ei, E, N, wb2, s2, xt2, out2, C2);

  // ---- pool + FC ----
  k_post2<<<nb((long long)N * C2), TB, 0, stream>>>(out2, b2, bat, pooled, cnt, N, C2);
  k_final<<<nb(G), TB, 0, stream>>>(pooled, cnt, fcw, fcb, outp, G, C2);
}
